// DMPNNEncoder_7619271983744
// MI455X (gfx1250) — matrix operand order repaired and checked by reading, not compiled, not run on hardware
//
#include <hip/hip_runtime.h>
#include <hip/hip_bf16.h>

// ---------------------------------------------------------------------------
// D-MPNN encoder for MI455X (gfx1250, wave32, WMMA bf16 16x16x32, f32 accum)
//   - weights staged in LDS per workgroup (ds_load B fragments)
//   - dual accumulator chains (2 n-tiles) for back-to-back WMMA issue
//   - message stored bf16 (WMMA quantizes A to bf16 anyway) -> 256MB stream
//   - e_sum (128MB f32) is L2-resident; scatter uses hw global_atomic_add_f32
// ---------------------------------------------------------------------------

typedef __attribute__((ext_vector_type(16))) __bf16 v16bf;
typedef __attribute__((ext_vector_type(4)))  __bf16 v4bf;
typedef __attribute__((ext_vector_type(8)))  float  v8f;
typedef __attribute__((ext_vector_type(8)))  unsigned int v8u;

#define DEVINL __device__ __forceinline__

constexpr int kAtoms = 250000;
constexpr int kEdges = 1000000;
constexpr int kOutD  = 128;
constexpr int kHalf  = kAtoms / 2;

// workspace layout (byte offsets from ws base)
constexpr size_t kWtIOff = 0;                                   // [128][192] bf16
constexpr size_t kWtHOff = kWtIOff + size_t(128) * 192 * 2;     // [128][128] bf16
constexpr size_t kWtOOff = kWtHOff + size_t(128) * 128 * 2;     // [128][256] bf16
constexpr size_t kWtMOff = kWtOOff + size_t(128) * 256 * 2;     // [128][128] bf16
constexpr size_t kWEnd   = kWtMOff + size_t(128) * 128 * 2;
constexpr size_t kInpOff = (kWEnd + 255) & ~size_t(255);        // inp  [E][128] f32
constexpr size_t kMsgOff = kInpOff + size_t(kEdges) * kOutD * 4;// msg  [E][128] bf16
constexpr size_t kSumOff = kMsgOff + size_t(kEdges) * kOutD * 2;// esum/h [N][128] f32

// -------------------------------- helpers ----------------------------------

DEVINL v8f zero8() {
    v8f z;
#pragma unroll
    for (int i = 0; i < 8; ++i) z[i] = 0.0f;
    return z;
}

// B fragment (32x16 bf16): lane holds col N=lane&15, K=(lane>>4)*8 + [0..7], 16+(lane>>4)*8 + [0..7].
// Weight panel is [N][Ktot] bf16, staged in LDS -> two 16B reads.
DEVINL v16bf ldB(const __bf16* Ws, int Ktot, int nt, int kt, int lane) {
    const __bf16* p = Ws + (size_t)(nt * 16 + (lane & 15)) * Ktot
                         + kt * 32 + (lane >> 4) * 8;
    uint4 lo = *(const uint4*)p, hi = *(const uint4*)(p + 16);
    v8u u = {lo.x, lo.y, lo.z, lo.w, hi.x, hi.y, hi.z, hi.w};
    return __builtin_bit_cast(v16bf, u);
}

// A fragment (16x32 bf16) from an f32 row pointer already offset to this
// lane's K-base: elements [0..7] = p[0..7], [8..15] = p[16..23].
DEVINL v16bf cvtA(const float* p) {
    v16bf r;
#pragma unroll
    for (int i = 0; i < 8; ++i) r[i] = (__bf16)p[i];
#pragma unroll
    for (int i = 0; i < 8; ++i) r[i + 8] = (__bf16)p[i + 16];
    return r;
}

DEVINL v8f bf16mma(v16bf a, v16bf b, v8f c) {
    return __builtin_amdgcn_wmma_f32_16x16x32_bf16(
        /*neg_a=*/false, a, /*neg_b=*/false, b,
        /*c_mod=*/(short)0, c, /*reuse_a=*/false, /*reuse_b=*/false);
}

// cooperative global->LDS weight copy (uint4 granularity), then barrier
DEVINL void stageW(uint4* sW, const char* ws, size_t off, int n16) {
    const uint4* g = (const uint4*)(ws + off);
    for (int i = threadIdx.x; i < n16; i += 256) sW[i] = g[i];
    __syncthreads();
}

// -------------------------- weight preparation -----------------------------
__global__ __launch_bounds__(256) void k_prep(
    const float* __restrict__ Wi, const float* __restrict__ Wh,
    const float* __restrict__ Wo, const float* __restrict__ Wm,
    char* __restrict__ ws) {
    __bf16* WtI = (__bf16*)(ws + kWtIOff);
    __bf16* WtH = (__bf16*)(ws + kWtHOff);
    __bf16* WtO = (__bf16*)(ws + kWtOOff);
    __bf16* WtM = (__bf16*)(ws + kWtMOff);
    int t = blockIdx.x * 256 + threadIdx.x;
    if (t < 192 * 128) { int k = t / 128, n = t % 128; WtI[n * 192 + k] = (__bf16)Wi[t]; return; }
    t -= 192 * 128;
    if (t < 128 * 128) { int k = t / 128, n = t % 128; WtH[n * 128 + k] = (__bf16)Wh[t]; return; }
    t -= 128 * 128;
    if (t < 256 * 128) { int k = t / 128, n = t % 128; WtO[n * 256 + k] = (__bf16)Wo[t]; return; }
    t -= 256 * 128;
    if (t < 128 * 128) { int k = t / 128, n = t % 128; WtM[n * 128 + k] = (__bf16)Wm[t]; }
}

__global__ __launch_bounds__(256) void k_zero(float4* __restrict__ p, int n4) {
    int i = blockIdx.x * 256 + threadIdx.x;
    if (i < n4) p[i] = make_float4(0.f, 0.f, 0.f, 0.f);
}

// -------- K1: inp = concat(atom[src], bond) @ W_i ; msg = relu(inp) --------
__global__ __launch_bounds__(256) void k_edge_init(
    const float* __restrict__ atom, const float* __restrict__ bond,
    const int* __restrict__ src, const char* __restrict__ ws) {
    __shared__ uint4 sW[192 * 128 * 2 / 16];            // 48 KB
    stageW(sW, ws, kWtIOff, 192 * 128 * 2 / 16);
    const __bf16* Ws = (const __bf16*)sW;

    float*  inp = (float*)(ws + kInpOff);
    __bf16* msg = (__bf16*)(ws + kMsgOff);

    const int lane = threadIdx.x & 31;
    const int tile = blockIdx.x * 8 + (threadIdx.x >> 5);
    if (tile >= kEdges / 16) return;
    const int e0  = tile * 16;
    const int e   = e0 + (lane & 15);
    const int s   = src[e];
    const int kb8 = (lane >> 4) * 8;

    v16bf A[6];
    const float* ap = atom + (size_t)s * 128;
#pragma unroll
    for (int kt = 0; kt < 4; ++kt) A[kt] = cvtA(ap + kt * 32 + kb8);
    const float* bp = bond + (size_t)e * 64;
#pragma unroll
    for (int kt = 0; kt < 2; ++kt) A[4 + kt] = cvtA(bp + kt * 32 + kb8);

#pragma unroll
    for (int np = 0; np < 4; ++np) {                    // 2 n-tiles per pass
        v8f a0 = zero8(), a1 = zero8();
#pragma unroll
        for (int kt = 0; kt < 6; ++kt) {
            v16bf b0 = ldB(Ws, 192, 2 * np,     kt, lane);
            v16bf b1 = ldB(Ws, 192, 2 * np + 1, kt, lane);
            a0 = bf16mma(A[kt], b0, a0);
            a1 = bf16mma(A[kt], b1, a1);
        }
        const int rbase = e0 + (lane >> 4) * 8;
        const int col0  = (2 * np) * 16 + (lane & 15);
#pragma unroll
        for (int v = 0; v < 8; ++v) {
            size_t i0 = (size_t)(rbase + v) * 128 + col0;
            float  x0 = a0[v], x1 = a1[v];
            inp[i0]      = x0;
            inp[i0 + 16] = x1;
            msg[i0]      = (__bf16)fmaxf(x0, 0.f);
            msg[i0 + 16] = (__bf16)fmaxf(x1, 0.f);
        }
    }
}

// -------- K2: e_sum[dst[e]] += msg[e]  (hardware f32 atomics, L2-resident) --
__global__ __launch_bounds__(256) void k_scatter(
    const int* __restrict__ dst, const char* __restrict__ ws) {
    const __bf16* msg  = (const __bf16*)(ws + kMsgOff);
    float*        esum = (float*)(ws + kSumOff);
    int t = blockIdx.x * 256 + threadIdx.x;      // E * 32 threads, 4 chans each
    int e = t >> 5;
    if (e >= kEdges) return;
    int c = (t & 31) * 4;
    v4bf m = *(const v4bf*)(msg + (size_t)e * 128 + c);
    float* o = esum + (size_t)dst[e] * 128 + c;
#pragma unroll
    for (int i = 0; i < 4; ++i) unsafeAtomicAdd(o + i, (float)m[i]);
}

// -------- K3: msg[e] = relu(inp[e] + (e_sum[src[e]] - msg[e^1]) @ W_h) -----
// e^1 lies in the same 16-edge tile -> in-place update is safe (A loads
// hoisted before stores; tiles disjoint across waves).
__global__ __launch_bounds__(256) void k_msg_update(
    const int* __restrict__ src, const char* __restrict__ ws) {
    __shared__ uint4 sW[128 * 128 * 2 / 16];            // 32 KB
    stageW(sW, ws, kWtHOff, 128 * 128 * 2 / 16);
    const __bf16* Ws = (const __bf16*)sW;

    const float* inp  = (const float*)(ws + kInpOff);
    const float* esum = (const float*)(ws + kSumOff);
    __bf16*      msg  = (__bf16*)(ws + kMsgOff);

    const int lane = threadIdx.x & 31;
    const int tile = blockIdx.x * 8 + (threadIdx.x >> 5);
    if (tile >= kEdges / 16) return;
    const int e0  = tile * 16;
    const int e   = e0 + (lane & 15);
    const int s   = src[e];       // == dst[e^1]
    const int kb8 = (lane >> 4) * 8;

    const float*  ep = esum + (size_t)s * 128;
    const __bf16* mp = msg + (size_t)(e ^ 1) * 128;
    v16bf A[4];
#pragma unroll
    for (int kt = 0; kt < 4; ++kt) {
        const int kb = kt * 32 + kb8;
        v16bf r;
#pragma unroll
        for (int i = 0; i < 8; ++i) r[i]     = (__bf16)(ep[kb + i]      - (float)mp[kb + i]);
#pragma unroll
        for (int i = 0; i < 8; ++i) r[i + 8] = (__bf16)(ep[kb + 16 + i] - (float)mp[kb + 16 + i]);
        A[kt] = r;
    }

#pragma unroll
    for (int np = 0; np < 4; ++np) {
        v8f a0 = zero8(), a1 = zero8();
#pragma unroll
        for (int kt = 0; kt < 4; ++kt) {
            v16bf b0 = ldB(Ws, 128, 2 * np,     kt, lane);
            v16bf b1 = ldB(Ws, 128, 2 * np + 1, kt, lane);
            a0 = bf16mma(A[kt], b0, a0);
            a1 = bf16mma(A[kt], b1, a1);
        }
        const int rbase = e0 + (lane >> 4) * 8;
        const int col0  = (2 * np) * 16 + (lane & 15);
#pragma unroll
        for (int v = 0; v < 8; ++v) {
            size_t i0 = (size_t)(rbase + v) * 128 + col0;
            msg[i0]      = (__bf16)fmaxf(inp[i0]      + a0[v], 0.f);
            msg[i0 + 16] = (__bf16)fmaxf(inp[i0 + 16] + a1[v], 0.f);
        }
    }
}

// -------- K4: h = relu(concat(atom, sum_ej) @ W_o + b_o), in-place in esum -
__global__ __launch_bounds__(256) void k_node_out(
    const float* __restrict__ atom, const float* __restrict__ b_o,
    char* __restrict__ ws) {
    __shared__ uint4 sW[256 * 128 * 2 / 16];            // 64 KB
    stageW(sW, ws, kWtOOff, 256 * 128 * 2 / 16);
    const __bf16* Ws = (const __bf16*)sW;

    float* esum = (float*)(ws + kSumOff);   // read sum_ej, write h (same rows)

    const int lane = threadIdx.x & 31;
    const int tile = blockIdx.x * 8 + (threadIdx.x >> 5);
    if (tile >= kAtoms / 16) return;
    const int n0  = tile * 16;
    const int n   = n0 + (lane & 15);
    const int kb8 = (lane >> 4) * 8;

    v16bf A[8];
    const float* ap = atom + (size_t)n * 128;
#pragma unroll
    for (int kt = 0; kt < 4; ++kt) A[kt] = cvtA(ap + kt * 32 + kb8);
    const float* sp = esum + (size_t)n * 128;
#pragma unroll
    for (int kt = 0; kt < 4; ++kt) A[4 + kt] = cvtA(sp + kt * 32 + kb8);

#pragma unroll
    for (int np = 0; np < 4; ++np) {
        v8f a0 = zero8(), a1 = zero8();
#pragma unroll
        for (int kt = 0; kt < 8; ++kt) {
            v16bf b0 = ldB(Ws, 256, 2 * np,     kt, lane);
            v16bf b1 = ldB(Ws, 256, 2 * np + 1, kt, lane);
            a0 = bf16mma(A[kt], b0, a0);
            a1 = bf16mma(A[kt], b1, a1);
        }
        const int col0  = (2 * np) * 16 + (lane & 15);
        const float bo0 = b_o[col0], bo1 = b_o[col0 + 16];
        const int rbase = n0 + (lane >> 4) * 8;
#pragma unroll
        for (int v = 0; v < 8; ++v) {
            size_t i0 = (size_t)(rbase + v) * 128 + col0;
            esum[i0]      = fmaxf(a0[v] + bo0, 0.f);
            esum[i0 + 16] = fmaxf(a1[v] + bo1, 0.f);
        }
    }
}

// -------- K5: out[seg[i]] += relu((h[i+HALF] - h[i]) @ W_mlp) --------------
__global__ __launch_bounds__(256) void k_rxn(
    const int* __restrict__ segids, const char* __restrict__ ws,
    float* __restrict__ out) {
    __shared__ uint4 sW[128 * 128 * 2 / 16];            // 32 KB
    stageW(sW, ws, kWtMOff, 128 * 128 * 2 / 16);
    const __bf16* Ws = (const __bf16*)sW;

    const float* h = (const float*)(ws + kSumOff);

    const int lane  = threadIdx.x & 31;
    const int tiles = (kHalf + 15) / 16;
    const int tile  = blockIdx.x * 8 + (threadIdx.x >> 5);
    if (tile >= tiles) return;
    const int r0  = tile * 16;
    const int r   = r0 + (lane & 15);
    const int rl  = r < kHalf ? r : kHalf - 1;   // clamp loads, mask stores
    const int kb8 = (lane >> 4) * 8;

    const float* p0 = h + (size_t)rl * 128;
    const float* p1 = h + (size_t)(rl + kHalf) * 128;
    v16bf A[4];
#pragma unroll
    for (int kt = 0; kt < 4; ++kt) {
        const int kb = kt * 32 + kb8;
        v16bf t;
#pragma unroll
        for (int i = 0; i < 8; ++i) t[i]     = (__bf16)(p1[kb + i]      - p0[kb + i]);
#pragma unroll
        for (int i = 0; i < 8; ++i) t[i + 8] = (__bf16)(p1[kb + 16 + i] - p0[kb + 16 + i]);
        A[kt] = t;
    }

#pragma unroll
    for (int np = 0; np < 4; ++np) {
        v8f a0 = zero8(), a1 = zero8();
#pragma unroll
        for (int kt = 0; kt < 4; ++kt) {
            v16bf b0 = ldB(Ws, 128, 2 * np,     kt, lane);
            v16bf b1 = ldB(Ws, 128, 2 * np + 1, kt, lane);
            a0 = bf16mma(A[kt], b0, a0);
            a1 = bf16mma(A[kt], b1, a1);
        }
        const int col0  = (2 * np) * 16 + (lane & 15);
        const int rbase = r0 + (lane >> 4) * 8;
#pragma unroll
        for (int v = 0; v < 8; ++v) {
            const int row = rbase + v;
            if (row < kHalf) {
                const int sg = segids[row];
                unsafeAtomicAdd(out + (size_t)sg * 128 + col0,      fmaxf(a0[v], 0.f));
                unsafeAtomicAdd(out + (size_t)sg * 128 + col0 + 16, fmaxf(a1[v], 0.f));
            }
        }
    }
}

// ------------------------------- launcher ----------------------------------

extern "C" void kernel_launch(void* const* d_in, const int* in_sizes, int n_in,
                              void* d_out, int out_size, void* d_ws, size_t ws_size,
                              hipStream_t stream) {
    const float* atom   = (const float*)d_in[0];
    const float* bond   = (const float*)d_in[1];
    const int*   src    = (const int*)d_in[2];
    const int*   dst    = (const int*)d_in[3];
    const int*   segids = (const int*)d_in[4];
    // d_in[5] = num_reactions (scalar, compile-time 1000 here)
    const float* Wi  = (const float*)d_in[6];
    const float* Wh  = (const float*)d_in[7];
    const float* Wo  = (const float*)d_in[8];
    const float* b_o = (const float*)d_in[9];
    const float* Wm  = (const float*)d_in[10];
    float* out = (float*)d_out;
    char*  ws  = (char*)d_ws;

    float4* esum4 = (float4*)(ws + kSumOff);
    const int esumN4 = kAtoms * kOutD / 4;            // 8,000,000
    const int outN4  = 1000 * kOutD / 4;              // 32,000

    const int edgeTiles = kEdges / 16;                // 62,500
    const int atomTiles = kAtoms / 16;                // 15,625
    const int diffTiles = (kHalf + 15) / 16;          // 7,813
    const int edgeBlks  = (edgeTiles + 7) / 8;
    const int atomBlks  = (atomTiles + 7) / 8;
    const int diffBlks  = (diffTiles + 7) / 8;

    // weights -> bf16 [N][K]
    k_prep<<<(90112 + 255) / 256, 256, 0, stream>>>(Wi, Wh, Wo, Wm, ws);

    // inp + initial message
    k_edge_init<<<edgeBlks, 256, 0, stream>>>(atom, bond, src, ws);

    // 4 message-passing iterations
    for (int d = 0; d < 4; ++d) {
        k_zero<<<(esumN4 + 255) / 256, 256, 0, stream>>>(esum4, esumN4);
        k_scatter<<<(kEdges * 32 + 255) / 256, 256, 0, stream>>>(dst, ws);
        k_msg_update<<<edgeBlks, 256, 0, stream>>>(src, ws);
    }

    // final sum_ej + node readout (h written in-place over e_sum)
    k_zero<<<(esumN4 + 255) / 256, 256, 0, stream>>>(esum4, esumN4);
    k_scatter<<<(kEdges * 32 + 255) / 256, 256, 0, stream>>>(dst, ws);
    k_node_out<<<atomBlks, 256, 0, stream>>>(atom, b_o, ws);

    // reaction head + pooling
    k_zero<<<(outN4 + 255) / 256, 256, 0, stream>>>((float4*)out, outN4);
    k_rxn<<<diffBlks, 256, 0, stream>>>(segids, ws, out);
}
